// MambaViTEncoder_86766929313981
// MI455X (gfx1250) — compile-verified, matrix operand order checked by reading, not run on hardware
//
#include <hip/hip_runtime.h>
#include <hip/hip_bf16.h>

#define D 768
#define HSZ 3072
#define SSZ 768
#define DEPTH 12
#define BATCH 8
#define TLEN 1024
#define M_TOT (BATCH * TLEN)   // 8192

typedef __attribute__((ext_vector_type(16))) __bf16 v16bf;
typedef __attribute__((ext_vector_type(8)))  float  v8f;

// ---------- helpers ----------
__device__ __forceinline__ unsigned short f2bf(float f) {
    unsigned int u = __float_as_uint(f);
    unsigned int r = (u + 0x7FFFu + ((u >> 16) & 1u)) >> 16;  // RNE
    return (unsigned short)r;
}
__device__ __forceinline__ float bf2f(unsigned short h) {
    return __uint_as_float(((unsigned int)h) << 16);
}
__device__ __forceinline__ void async_wait0() {
    asm volatile("s_wait_asynccnt 0x0" ::: "memory");
}

// ---------- fp32 -> bf16 convert (elementwise, A-side activations) ----------
__global__ void cvt_bf16_kernel(const float* __restrict__ src,
                                unsigned short* __restrict__ dst, int n) {
    int i = blockIdx.x * blockDim.x + threadIdx.x;
    if (i < n) dst[i] = f2bf(src[i]);
}

// ---------- fp32 -> bf16 convert + transpose: src (R x C) -> dst (C x R) ----------
__global__ __launch_bounds__(256) void cvt_t_kernel(const float* __restrict__ src,
                                                    unsigned short* __restrict__ dst,
                                                    int R, int C) {
    __shared__ float tile[32][33];
    int c0 = blockIdx.x * 32, r0 = blockIdx.y * 32;
    int tx = threadIdx.x, ty = threadIdx.y;   // 32 x 8
#pragma unroll
    for (int j = 0; j < 32; j += 8)
        tile[ty + j][tx] = src[(size_t)(r0 + ty + j) * C + c0 + tx];
    __syncthreads();
#pragma unroll
    for (int j = 0; j < 32; j += 8)
        dst[(size_t)(c0 + ty + j) * R + r0 + tx] = f2bf(tile[tx][ty + j]);
}

// ---------- per-sample coordinate min ----------
__global__ void coordmin_kernel(const float* __restrict__ coords, float* __restrict__ ref) {
    int b = blockIdx.x;
    __shared__ float sx[256], sy[256];
    float mx = 3.4e38f, my = 3.4e38f;
    for (int t = threadIdx.x; t < TLEN; t += 256) {
        mx = fminf(mx, coords[(size_t)(b * TLEN + t) * 2 + 0]);
        my = fminf(my, coords[(size_t)(b * TLEN + t) * 2 + 1]);
    }
    sx[threadIdx.x] = mx; sy[threadIdx.x] = my;
    __syncthreads();
    for (int s = 128; s > 0; s >>= 1) {
        if (threadIdx.x < s) {
            sx[threadIdx.x] = fminf(sx[threadIdx.x], sx[threadIdx.x + s]);
            sy[threadIdx.x] = fminf(sy[threadIdx.x], sy[threadIdx.x + s]);
        }
        __syncthreads();
    }
    if (threadIdx.x == 0) { ref[b * 2 + 0] = sx[0]; ref[b * 2 + 1] = sy[0]; }
}

// ---------- sinusoidal positional embedding add (in place on X) ----------
__global__ void posadd_kernel(const float* __restrict__ coords,
                              const float* __restrict__ ref,
                              float* __restrict__ X) {
    int i = blockIdx.x * blockDim.x + threadIdx.x;
    if (i >= M_TOT * D) return;
    int m = i / D, d = i % D;
    int b = m / TLEN;
    int axis = d / 384;
    int j = d % 384;
    int fi = (j < 192) ? j : (j - 192);
    float freq = __expf(-9.210340371976184f * (float)fi / 192.0f);  // 10000^{-fi/192}
    float rel = (coords[(size_t)m * 2 + axis] - ref[b * 2 + axis]) * (1.0f / 128.0f);
    float ang = rel * freq;
    X[i] += (j < 192) ? __sinf(ang) : __cosf(ang);
}

// ---------- LayerNorm (row per block), bf16 or f32 output ----------
template <int OUT_BF16>
__global__ __launch_bounds__(256) void layernorm_kernel(
    const float* __restrict__ x, const float* __restrict__ g, const float* __restrict__ b,
    unsigned short* __restrict__ outh, float* __restrict__ outf) {
    int row = blockIdx.x;
    const float* xr = x + (size_t)row * D;
    __shared__ float s1[256], s2[256];
    float v[3], sum = 0.f, sq = 0.f;
#pragma unroll
    for (int j = 0; j < 3; j++) {
        v[j] = xr[threadIdx.x + j * 256];
        sum += v[j]; sq += v[j] * v[j];
    }
    s1[threadIdx.x] = sum; s2[threadIdx.x] = sq;
    __syncthreads();
    for (int s = 128; s > 0; s >>= 1) {
        if (threadIdx.x < s) {
            s1[threadIdx.x] += s1[threadIdx.x + s];
            s2[threadIdx.x] += s2[threadIdx.x + s];
        }
        __syncthreads();
    }
    float mean = s1[0] * (1.0f / D);
    float var  = s2[0] * (1.0f / D) - mean * mean;
    float rstd = rsqrtf(var + 1e-5f);
#pragma unroll
    for (int j = 0; j < 3; j++) {
        int d = threadIdx.x + j * 256;
        float o = (v[j] - mean) * rstd * g[d] + b[d];
        if (OUT_BF16) outh[(size_t)row * D + d] = f2bf(o);
        else          outf[(size_t)row * D + d] = o;
    }
}

// ---------- gate kernel ----------
__global__ void gates_kernel(const float* __restrict__ go,
                             const unsigned short* __restrict__ xn,
                             unsigned short* __restrict__ xg,
                             unsigned short* __restrict__ g2) {
    int i = blockIdx.x * blockDim.x + threadIdx.x;
    if (i >= M_TOT * D) return;
    int m = i / D, d = i % D;
    float a = go[(size_t)m * (2 * D) + d];
    float c = go[(size_t)m * (2 * D) + D + d];
    float sa = a * (1.0f / (1.0f + __expf(-a)));
    float sc = c * (1.0f / (1.0f + __expf(-c)));
    xg[i] = f2bf(bf2f(xn[i]) * sa);
    g2[i] = f2bf(sc);
}

// ---------- delta kernel: d = softplus(rowmean(sd)), also stash exp(-d) ----------
__global__ __launch_bounds__(256) void delta_kernel(const float* __restrict__ sd,
                                                    float* __restrict__ delta,
                                                    float* __restrict__ dexp) {
    int row = blockIdx.x;
    const float* xr = sd + (size_t)row * SSZ;
    __shared__ float s1[256];
    float sum = 0.f;
#pragma unroll
    for (int j = 0; j < 3; j++) sum += xr[threadIdx.x + j * 256];
    s1[threadIdx.x] = sum;
    __syncthreads();
    for (int s = 128; s > 0; s >>= 1) {
        if (threadIdx.x < s) s1[threadIdx.x] += s1[threadIdx.x + s];
        __syncthreads();
    }
    if (threadIdx.x == 0) {
        float x = s1[0] * (1.0f / SSZ);
        float dlt = (x > 20.0f) ? x : log1pf(__expf(x));
        delta[row] = dlt;
        dexp[row]  = __expf(-dlt);
    }
}

// ---------- selective scan ----------
__global__ void scan_kernel(float* __restrict__ u, const float* __restrict__ bm,
                            const float* __restrict__ delta, const float* __restrict__ dexp) {
    int gid = blockIdx.x * blockDim.x + threadIdx.x;   // over B*S = 6144
    if (gid >= BATCH * SSZ) return;
    int b = gid / SSZ, s = gid % SSZ;
    float h = 0.0f;
#pragma unroll 4
    for (int t = 0; t < TLEN; t++) {
        int r = b * TLEN + t;
        float d  = delta[r];
        float da = dexp[r];
        size_t idx = (size_t)r * SSZ + s;
        h = da * h + d * bm[idx] * u[idx];
        u[idx] = h;
    }
}

// ---------- bf16 WMMA GEMM ----------
// C = A(MxK) * Bt^T + bias ; Bt is the PRE-TRANSPOSED weight, laid out N x K (bf16).
// block = 256 threads = 8 wave32 waves; block tile 256(M) x 128(N); wave tile 64x64 (4x4 accs).
// LDS double-buffered; tiles staged with GLOBAL_LOAD_ASYNC_TO_LDS_B128 (ASYNCcnt),
// one barrier per K-step of 32.
// EPI: 0 = f32 out = acc+bias ; 1 = f32 out = acc+bias+resid ; 2 = bf16 out = gelu(acc+bias)
template <int EPI>
__global__ __launch_bounds__(256) void gemm_bf16_kernel(
    const unsigned short* __restrict__ A,    // M x K  bf16
    const unsigned short* __restrict__ Bt,   // N x K  bf16 (transposed weight)
    const float* __restrict__ bias,          // N
    float* outf,                             // f32 out (EPI 0/1)
    unsigned short* outh,                    // bf16 out (EPI 2)
    const float* resid,                      // residual (EPI 1; may alias outf)
    int M, int N, int K) {
    __shared__ __align__(16) unsigned short lA[2][256][40];
    __shared__ __align__(16) unsigned short lB[2][128][40];

    const int tid  = threadIdx.x;
    const int wave = tid >> 5;
    const int lane = tid & 31;
    const int bm   = blockIdx.y * 256;
    const int bn   = blockIdx.x * 128;
    const int wm   = (wave & 3) * 64;     // wave m-offset in tile
    const int wn   = (wave >> 2) * 64;    // wave n-offset in tile
    const int lhalf = lane >> 4;          // 0/1
    const int lmod  = lane & 15;

    v8f acc[4][4];
#pragma unroll
    for (int i = 0; i < 4; i++)
#pragma unroll
        for (int j = 0; j < 4; j++)
#pragma unroll
            for (int r = 0; r < 8; r++) acc[i][j][r] = 0.0f;

    // cooperative-load assignments (both tiles are row-major-in-K)
    const int brow  = tid >> 1;   // 0..127 (B tile row = n)
    const int bhalf = tid & 1;    // 16-elem half

    union FU { uint4 q[2]; v16bf v; };

    // per-thread LDS byte addresses (low 32 bits of generic LDS pointer = LDS offset)
    const unsigned ldsA0 = (unsigned)(size_t)&lA[0][tid][0];
    const unsigned ldsA1 = (unsigned)(size_t)&lA[1][tid][0];
    const unsigned ldsB0 = (unsigned)(size_t)&lB[0][brow][bhalf * 16];
    const unsigned ldsB1 = (unsigned)(size_t)&lB[1][brow][bhalf * 16];
    // per-thread global byte offsets (GVS mode: saddr_u64 + vaddr_i32)
    const unsigned gofA = (unsigned)((size_t)(bm + tid) * K * 2);
    const unsigned gofB = (unsigned)(((size_t)(bn + brow) * K + bhalf * 16) * 2);

    // async-stage the 256x32 A tile (64B/thread) and 128x32 B tile (32B/thread)
    auto stage = [&](int buf, int k0) {
        unsigned la = buf ? ldsA1 : ldsA0;
        unsigned lb = buf ? ldsB1 : ldsB0;
        unsigned oa = gofA + (unsigned)(k0 * 2);
        unsigned ob = gofB + (unsigned)(k0 * 2);
        asm volatile(
            "global_load_async_to_lds_b128 %0, %1, %2 offset:0\n\t"
            "global_load_async_to_lds_b128 %0, %1, %2 offset:16\n\t"
            "global_load_async_to_lds_b128 %0, %1, %2 offset:32\n\t"
            "global_load_async_to_lds_b128 %0, %1, %2 offset:48"
            :: "v"(la), "v"(oa), "s"((unsigned long long)A) : "memory");
        asm volatile(
            "global_load_async_to_lds_b128 %0, %1, %2 offset:0\n\t"
            "global_load_async_to_lds_b128 %0, %1, %2 offset:16"
            :: "v"(lb), "v"(ob), "s"((unsigned long long)Bt) : "memory");
    };

    stage(0, 0);
    async_wait0();
    __syncthreads();
    int cur = 0;

    for (int k0 = 0; k0 < K; k0 += 32) {
        if (k0 + 32 < K) stage(cur ^ 1, k0 + 32);   // async copy overlaps WMMAs below

        FU af, bfr[4];
#pragma unroll
        for (int ni = 0; ni < 4; ni++) {
            const unsigned short* p = &lB[cur][wn + ni * 16 + lmod][0];
            bfr[ni].q[0] = *(const uint4*)(p + lhalf * 8);
            bfr[ni].q[1] = *(const uint4*)(p + 16 + lhalf * 8);
        }
#pragma unroll
        for (int mi = 0; mi < 4; mi++) {
            const unsigned short* p = &lA[cur][wm + mi * 16 + lmod][0];
            af.q[0] = *(const uint4*)(p + lhalf * 8);        // K = half*8 .. +7
            af.q[1] = *(const uint4*)(p + 16 + lhalf * 8);   // K = 16+half*8 .. +7
#pragma unroll
            for (int ni = 0; ni < 4; ni++)
                acc[mi][ni] = __builtin_amdgcn_wmma_f32_16x16x32_bf16(
                    false, af.v, false, bfr[ni].v, (short)0, acc[mi][ni], false, false);
        }
        async_wait0();       // own async copies into nxt done
        __syncthreads();     // everyone's copies done + reads of cur done
        cur ^= 1;
    }

    // branch-free epilogue (per template instantiation)
#pragma unroll
    for (int mi = 0; mi < 4; mi++) {
#pragma unroll
        for (int ni = 0; ni < 4; ni++) {
            int n_g = bn + wn + ni * 16 + lmod;
            float bv = bias[n_g];
#pragma unroll
            for (int r = 0; r < 8; r++) {
                int m_g = bm + wm + mi * 16 + lhalf * 8 + r;
                size_t o = (size_t)m_g * N + n_g;
                float v = acc[mi][ni][r] + bv;
                if (EPI == 1) v += resid[o];
                if (EPI == 2) {
                    float gv = 0.5f * v * (1.0f + erff(v * 0.7071067811865475f));
                    outh[o] = f2bf(gv);
                } else {
                    outf[o] = v;
                }
            }
        }
    }
}

// ---------- orchestration ----------
static inline size_t alup(size_t x) { return (x + 255) & ~(size_t)255; }

extern "C" void kernel_launch(void* const* d_in, const int* in_sizes, int n_in,
                              void* d_out, int out_size, void* d_ws, size_t ws_size,
                              hipStream_t stream) {
    const float* tokens = (const float*)d_in[0];
    const float* coords = (const float*)d_in[1];
    const float* W_tok  = (const float*)d_in[2];
    const float* b_tok  = (const float*)d_in[3];
    const float* ln1_g  = (const float*)d_in[4];
    const float* ln1_b  = (const float*)d_in[5];
    const float* Wg     = (const float*)d_in[6];
    const float* bg     = (const float*)d_in[7];
    const float* Ws     = (const float*)d_in[8];
    const float* bs     = (const float*)d_in[9];
    const float* Wo     = (const float*)d_in[10];
    const float* bo     = (const float*)d_in[11];
    const float* ln2_g  = (const float*)d_in[12];
    const float* ln2_b  = (const float*)d_in[13];
    const float* W1     = (const float*)d_in[14];
    const float* b1     = (const float*)d_in[15];
    const float* W2     = (const float*)d_in[16];
    const float* b2     = (const float*)d_in[17];
    const float* lnf_g  = (const float*)d_in[18];
    const float* lnf_b  = (const float*)d_in[19];

    char* ws = (char*)d_ws;
    size_t off = 0;
    float* X            = (float*)(ws + off);          off += alup((size_t)M_TOT * D * 4);
    // contiguous bf16 triple: [XG | G2 | A16] so the three Ws-GEMMs fuse into one
    unsigned short* XG  = (unsigned short*)(ws + off); off += (size_t)M_TOT * D * 2;
    unsigned short* G2  = (unsigned short*)(ws + off); off += (size_t)M_TOT * D * 2;
    unsigned short* A16 = (unsigned short*)(ws + off); off += (size_t)M_TOT * D * 2;
    // contiguous f32 triple: [U | BM | SD]
    float* U            = (float*)(ws + off);          off += (size_t)M_TOT * SSZ * 4;
    float* BM           = (float*)(ws + off);          off += (size_t)M_TOT * SSZ * 4;
    float* SD           = (float*)(ws + off);          off += (size_t)M_TOT * SSZ * 4;
    float* GO           = (float*)(ws + off);          off += alup((size_t)M_TOT * 2 * D * 4); // aliased as H16 (bf16 MxH)
    float* DELTA        = (float*)(ws + off);          off += alup((size_t)M_TOT * 4);
    float* DEXP         = (float*)(ws + off);          off += alup((size_t)M_TOT * 4);
    unsigned short* WB  = (unsigned short*)(ws + off); off += alup((size_t)D * HSZ * 2);        // transposed weight scratch
    float* REF          = (float*)(ws + off);          off += alup(64);
    unsigned short* H16 = (unsigned short*)GO;

    const int M = M_TOT;
    dim3 blk(256);
    dim3 tblk(32, 8);
    auto nblk = [](int n) { return dim3((n + 255) / 256); };
    auto tgrid = [](int R, int C) { return dim3(C / 32, R / 32); };  // for cvt_t (src RxC)

    // embedding: X = tokens @ W_tok + b_tok + pos_embed(coords)
    cvt_bf16_kernel<<<nblk(M * D), blk, 0, stream>>>(tokens, A16, M * D);
    cvt_t_kernel<<<tgrid(D, D), tblk, 0, stream>>>(W_tok, WB, D, D);
    gemm_bf16_kernel<0><<<dim3(D / 128, M / 256), blk, 0, stream>>>(
        A16, WB, b_tok, X, nullptr, nullptr, M, D, D);
    coordmin_kernel<<<dim3(BATCH), blk, 0, stream>>>(coords, REF);
    posadd_kernel<<<nblk(M * D), blk, 0, stream>>>(coords, REF, X);

    for (int l = 0; l < DEPTH; l++) {
        const float* Wg_l = Wg + (size_t)l * D * 2 * D;
        const float* bg_l = bg + (size_t)l * 2 * D;
        const float* Ws_l = Ws + (size_t)l * D * SSZ;
        const float* bs_l = bs + (size_t)l * SSZ;
        const float* Wo_l = Wo + (size_t)l * SSZ * D;
        const float* bo_l = bo + (size_t)l * D;
        const float* W1_l = W1 + (size_t)l * D * HSZ;
        const float* b1_l = b1 + (size_t)l * HSZ;
        const float* W2_l = W2 + (size_t)l * HSZ * D;
        const float* b2_l = b2 + (size_t)l * D;

        // xn = LN1(x) -> bf16
        layernorm_kernel<1><<<dim3(M), blk, 0, stream>>>(X, ln1_g + l * D, ln1_b + l * D, A16, nullptr);
        // go = xn @ Wg + bg
        cvt_t_kernel<<<tgrid(D, 2 * D), tblk, 0, stream>>>(Wg_l, WB, D, 2 * D);
        gemm_bf16_kernel<0><<<dim3(2 * D / 128, M / 256), blk, 0, stream>>>(
            A16, WB, bg_l, GO, nullptr, nullptr, M, 2 * D, D);
        // gates
        gates_kernel<<<nblk(M * D), blk, 0, stream>>>(GO, A16, XG, G2);
        // fused triple GEMM against Ws: [XG;G2;A16] @ Ws -> [U;BM;SD]
        cvt_t_kernel<<<tgrid(D, SSZ), tblk, 0, stream>>>(Ws_l, WB, D, SSZ);
        gemm_bf16_kernel<0><<<dim3(SSZ / 128, 3 * M / 256), blk, 0, stream>>>(
            XG, WB, bs_l, U, nullptr, nullptr, 3 * M, SSZ, D);
        // delta = softplus(rowmean(SD)); also exp(-delta)
        delta_kernel<<<dim3(M), blk, 0, stream>>>(SD, DELTA, DEXP);
        // selective scan (y in-place over U)
        scan_kernel<<<dim3((BATCH * SSZ + 255) / 256), blk, 0, stream>>>(U, BM, DELTA, DEXP);
        // x += y @ Wo + bo
        cvt_bf16_kernel<<<nblk(M * SSZ), blk, 0, stream>>>(U, A16, M * SSZ);
        cvt_t_kernel<<<tgrid(SSZ, D), tblk, 0, stream>>>(Wo_l, WB, SSZ, D);
        gemm_bf16_kernel<1><<<dim3(D / 128, M / 256), blk, 0, stream>>>(
            A16, WB, bo_l, X, nullptr, X, M, D, SSZ);
        // MLP: h = gelu(LN2(x) @ W1 + b1) ; x += h @ W2 + b2
        layernorm_kernel<1><<<dim3(M), blk, 0, stream>>>(X, ln2_g + l * D, ln2_b + l * D, A16, nullptr);
        cvt_t_kernel<<<tgrid(D, HSZ), tblk, 0, stream>>>(W1_l, WB, D, HSZ);
        gemm_bf16_kernel<2><<<dim3(HSZ / 128, M / 256), blk, 0, stream>>>(
            A16, WB, b1_l, nullptr, H16, nullptr, M, HSZ, D);
        cvt_t_kernel<<<tgrid(HSZ, D), tblk, 0, stream>>>(W2_l, WB, HSZ, D);
        gemm_bf16_kernel<1><<<dim3(D / 128, M / 256), blk, 0, stream>>>(
            H16, WB, b2_l, X, nullptr, X, M, D, HSZ);
    }

    // final layernorm -> output (f32)
    layernorm_kernel<0><<<dim3(M), blk, 0, stream>>>(X, lnf_g, lnf_b, nullptr, (float*)d_out);
}
